// SelfAttention_Realformer_24026047054590
// MI455X (gfx1250) — matrix operand order repaired and checked by reading, not compiled, not run on hardware
//
#include <hip/hip_runtime.h>

// ---------------------------------------------------------------------------
// RealFormer self-attention forward for gfx1250 (MI455X), bf16 WMMA pipeline.
// Staging path uses GLOBAL_LOAD_ASYNC_TO_LDS_B128 (ASYNCcnt) — builtin is
// present on this toolchain with signature (v4i AS1*, v4i AS3*, imm, imm).
// ---------------------------------------------------------------------------

typedef __attribute__((ext_vector_type(16))) __bf16       v16bf;
typedef __attribute__((ext_vector_type(8)))  float        v8f;
typedef __attribute__((ext_vector_type(4)))  unsigned int v4u;
typedef __attribute__((ext_vector_type(8)))  unsigned int v8u;
typedef int v4i __attribute__((vector_size(4 * sizeof(int))));
typedef __attribute__((address_space(1))) v4i* gptr_v4i;
typedef __attribute__((address_space(3))) v4i* lptr_v4i;

#define DEVINL __device__ __forceinline__

static constexpr int B_  = 4;
static constexpr int L_  = 1024;
static constexpr int H_  = 12;
static constexpr int DK_ = 64;
static constexpr int F_  = 768;        // FEA == H_*DK_
static constexpr int M_  = B_ * L_;    // 4096 rows in projection GEMMs
static constexpr float MASKV = -32767.0f;   // -2^15 + 1

#if defined(__has_builtin)
#if __has_builtin(__builtin_amdgcn_global_load_async_to_lds_b128)
#define HAVE_ASYNC_LDS 1
#endif
#endif

DEVINL unsigned short f2bf(float f) {
  unsigned int u = __builtin_bit_cast(unsigned int, f);
  u += 0x7fffu + ((u >> 16) & 1u);     // round-to-nearest-even
  return (unsigned short)(u >> 16);
}

#ifdef HAVE_ASYNC_LDS
// Per-lane 16B direct global->LDS async copy (tracked with ASYNCcnt).
DEVINL void async_cp16(const unsigned short* g, unsigned short* l) {
  __builtin_amdgcn_global_load_async_to_lds_b128(
      (gptr_v4i)(v4i*)g, (lptr_v4i)(v4i*)l, 0, 0);
}
DEVINL void wait_async0() {
#if __has_builtin(__builtin_amdgcn_s_wait_asynccnt)
  __builtin_amdgcn_s_wait_asynccnt(0);
#else
  asm volatile("s_wait_asynccnt 0x0" ::: "memory");
#endif
}
#endif

// ------------------------- WMMA fragment loaders ---------------------------
// LDS tiles are bf16 (ushort) with padded row stride LDSS (96B -> every
// 16B chunk we touch is 16B aligned for ds_load_b128).
static constexpr int LDSS = 48;

// A operand, 16x32 bf16 (ISA 7.12.2): lanes 0-15 hold M=lane, K{0-7,16-23};
// lanes 16-31 hold M=lane-16, K{8-15,24-31}.
DEVINL v16bf load_frag_a(const unsigned short* tile, int lane) {
  const unsigned short* r = tile + (lane & 15) * LDSS + ((lane >> 4) << 3);
  v4u lo = *reinterpret_cast<const v4u*>(r);
  v4u hi = *reinterpret_cast<const v4u*>(r + 16);
  v8u u;
  u[0] = lo[0]; u[1] = lo[1]; u[2] = lo[2]; u[3] = lo[3];
  u[4] = hi[0]; u[5] = hi[1]; u[6] = hi[2]; u[7] = hi[3];
  return __builtin_bit_cast(v16bf, u);
}

// B operand, 32x16 bf16 stored N-major in LDS (16 rows of K=32): lane n holds
// column n; lanes 0-15 K{0-7,16-23}, lanes 16-31 K{8-15,24-31}.
DEVINL v16bf load_frag_b(const unsigned short* tile, int lane) {
  const unsigned short* r = tile + (lane & 15) * LDSS + ((lane >> 4) << 3);
  v4u lo = *reinterpret_cast<const v4u*>(r);
  v4u hi = *reinterpret_cast<const v4u*>(r + 16);
  v8u u;
  u[0] = lo[0]; u[1] = lo[1]; u[2] = lo[2]; u[3] = lo[3];
  u[4] = hi[0]; u[5] = hi[1]; u[6] = hi[2]; u[7] = hi[3];
  return __builtin_bit_cast(v16bf, u);
}

// ------------------------- 128x64 block GEMM core --------------------------
// C(128x64) = A(128xK) * B(64xK)^T, A/B bf16 row-major, f32 accumulate.
// 256 threads = 8 waves arranged 4(M) x 2(N); each wave: 2x2 WMMA tiles.
template <typename Epi>
DEVINL void gemm_tile(const unsigned short* __restrict__ A, int lda,
                      const unsigned short* __restrict__ B, int ldb,
                      int K, unsigned short* As, unsigned short* Bs, Epi epi) {
  const int tid  = threadIdx.x;
  const int lane = tid & 31;
  const int wave = tid >> 5;
  const int wm   = (wave >> 1) << 5;   // 0,32,64,96
  const int wn   = (wave & 1) << 5;    // 0,32

  v8f acc[2][2] = {};
  for (int k0 = 0; k0 < K; k0 += 32) {
#ifdef HAVE_ASYNC_LDS
    // Direct global->LDS async copies: A tile 512 chunks, B tile 256 chunks.
#pragma unroll
    for (int i = 0; i < 2; ++i) {
      int c = tid + (i << 8);
      int row = c >> 2, kc = (c & 3) << 3;
      async_cp16(&A[(size_t)row * lda + k0 + kc], &As[row * LDSS + kc]);
    }
    {
      int row = tid >> 2, kc = (tid & 3) << 3;
      async_cp16(&B[(size_t)row * ldb + k0 + kc], &Bs[row * LDSS + kc]);
    }
#else
    // Stage A tile (128x32): 512 chunks of 8 ushorts, 2 per thread.
#pragma unroll
    for (int i = 0; i < 2; ++i) {
      int c = tid + (i << 8);
      int row = c >> 2, kc = (c & 3) << 3;
      *reinterpret_cast<v4u*>(&As[row * LDSS + kc]) =
          *reinterpret_cast<const v4u*>(&A[(size_t)row * lda + k0 + kc]);
    }
    // Stage B tile (64x32): 256 chunks, 1 per thread.
    {
      int row = tid >> 2, kc = (tid & 3) << 3;
      *reinterpret_cast<v4u*>(&Bs[row * LDSS + kc]) =
          *reinterpret_cast<const v4u*>(&B[(size_t)row * ldb + k0 + kc]);
    }
#endif
    if (k0 + 32 < K) {  // speculative prefetch of next K-step -> global_prefetch_b8
      __builtin_prefetch(&A[(size_t)(tid >> 1) * lda + k0 + 32]);
      __builtin_prefetch(&B[(size_t)(tid & 63) * ldb + k0 + 32]);
    }
#ifdef HAVE_ASYNC_LDS
    wait_async0();
#endif
    __syncthreads();

    v16bf af[2], bfr[2];
#pragma unroll
    for (int i = 0; i < 2; ++i) af[i]  = load_frag_a(&As[(wm + (i << 4)) * LDSS], lane);
#pragma unroll
    for (int j = 0; j < 2; ++j) bfr[j] = load_frag_b(&Bs[(wn + (j << 4)) * LDSS], lane);

#pragma unroll
    for (int i = 0; i < 2; ++i)
#pragma unroll
      for (int j = 0; j < 2; ++j)
        acc[i][j] = __builtin_amdgcn_wmma_f32_16x16x32_bf16(
            false, af[i], false, bfr[j], (short)0, acc[i][j], false, false);
    __syncthreads();
  }
#pragma unroll
  for (int i = 0; i < 2; ++i)
#pragma unroll
    for (int j = 0; j < 2; ++j) epi(wm + (i << 4), wn + (j << 4), acc[i][j]);
}

// C/D f32 16x16 layout: VGPR v, lanes 0-15 -> (M=v, N=lane); lanes 16-31 ->
// (M=v+8, N=lane-16). Epilogues compute row0 = mt + ((lane>>4)<<3) + v.

// ------------------------------ kernels ------------------------------------

__global__ void __launch_bounds__(256)
cvt_kernel(const float* __restrict__ src, unsigned short* __restrict__ dst, int n) {
  int i = blockIdx.x * 256 + threadIdx.x;
  if (i < n) dst[i] = f2bf(src[i]);
}

// X(4096x768) @ W(768x768)^T + bias -> bf16, layout (B,H,L,DK) or (B,H,DK,L).
__global__ void __launch_bounds__(256)
proj_kernel(const unsigned short* __restrict__ X, const unsigned short* __restrict__ W,
            const float* __restrict__ bias, unsigned short* __restrict__ dst,
            int transposed) {
  __shared__ alignas(16) unsigned short As[128 * LDSS];
  __shared__ alignas(16) unsigned short Bs[64 * LDSS];
  const int m0 = blockIdx.x << 7;
  const int n0 = blockIdx.y << 6;
  const unsigned short* A  = X + (size_t)m0 * F_;
  const unsigned short* Bp = W + (size_t)n0 * F_;
  auto epi = [=](int mt, int nt, const v8f& a) {
    const int lane = threadIdx.x & 31;
    const int row0 = m0 + mt + ((lane >> 4) << 3);
    const int col  = n0 + nt + (lane & 15);
    const int h = col >> 6, dd = col & 63;
    const float bv = bias[col];
#pragma unroll
    for (int v = 0; v < 8; ++v) {
      const int m = row0 + v;
      const int b = m >> 10, l = m & 1023;
      const size_t idx = transposed
          ? ((size_t)((b * H_ + h) * DK_ + dd)) * L_ + l
          : ((size_t)((b * H_ + h) * L_ + l)) * DK_ + dd;
      dst[idx] = f2bf(a[v] + bv);
    }
  };
  gemm_tile(A, F_, Bp, F_, F_, As, Bs, epi);
}

// scores = Q K^T / 8 + preScores, masked; f32 straight into d_out region 2.
__global__ void __launch_bounds__(256)
qk_kernel(const unsigned short* __restrict__ Q, const unsigned short* __restrict__ Km,
          const float* __restrict__ pre, const int* __restrict__ mask,
          float* __restrict__ scores) {
  __shared__ alignas(16) unsigned short As[128 * LDSS];
  __shared__ alignas(16) unsigned short Bs[64 * LDSS];
  const int z  = blockIdx.z;           // b*H + h
  const int b  = z / H_;
  const int m0 = blockIdx.x << 7;
  const int n0 = blockIdx.y << 6;
  const unsigned short* A  = Q  + ((size_t)z * L_ + m0) * DK_;
  const unsigned short* Bp = Km + ((size_t)z * L_ + n0) * DK_;
  const float* prez = pre + (size_t)z * L_ * L_;
  float* sz = scores + (size_t)z * L_ * L_;
  const int* mb = mask + (size_t)b * L_ * L_;
  auto epi = [=](int mt, int nt, const v8f& a) {
    const int lane = threadIdx.x & 31;
    const int row0 = m0 + mt + ((lane >> 4) << 3);
    const int col  = n0 + nt + (lane & 15);
#pragma unroll
    for (int v = 0; v < 8; ++v) {
      const int r = row0 + v;
      float s = a[v] * 0.125f + prez[(size_t)r * L_ + col];
      if (mb[(size_t)r * L_ + col] == 0) s = MASKV;
      sz[(size_t)r * L_ + col] = s;
    }
  };
  gemm_tile(A, DK_, Bp, DK_, DK_, As, Bs, epi);
}

// Row softmax over L=1024, f32 scores -> bf16 alpha. One block per row.
__global__ void __launch_bounds__(256)
softmax_kernel(const float* __restrict__ scores, unsigned short* __restrict__ alpha) {
  const size_t row = blockIdx.x;
  const float* s = scores + row * L_;
  unsigned short* a = alpha + row * L_;
  const int tid = threadIdx.x;
  __shared__ float redm[8];
  __shared__ float reds[8];

  float vals[4];
  float lm = -3.4e38f;
#pragma unroll
  for (int i = 0; i < 4; ++i) { vals[i] = s[tid + 256 * i]; lm = fmaxf(lm, vals[i]); }
#pragma unroll
  for (int off = 16; off > 0; off >>= 1) lm = fmaxf(lm, __shfl_xor(lm, off, 32));
  if ((tid & 31) == 0) redm[tid >> 5] = lm;
  __syncthreads();
  float mx = redm[0];
#pragma unroll
  for (int i = 1; i < 8; ++i) mx = fmaxf(mx, redm[i]);

  float e[4];
  float ls = 0.f;
#pragma unroll
  for (int i = 0; i < 4; ++i) { e[i] = __expf(vals[i] - mx); ls += e[i]; }
#pragma unroll
  for (int off = 16; off > 0; off >>= 1) ls += __shfl_xor(ls, off, 32);
  if ((tid & 31) == 0) reds[tid >> 5] = ls;
  __syncthreads();
  float sum = 0.f;
#pragma unroll
  for (int i = 0; i < 8; ++i) sum += reds[i];
  const float inv = 1.0f / sum;
#pragma unroll
  for (int i = 0; i < 4; ++i) a[tid + 256 * i] = f2bf(e[i] * inv);
}

// z = alpha @ V (V kept transposed (B,H,DK,L) so B operand is K-contiguous);
// result stored bf16 as (B, L, H*DK) for the final projection.
__global__ void __launch_bounds__(256)
av_kernel(const unsigned short* __restrict__ alpha, const unsigned short* __restrict__ VT,
          unsigned short* __restrict__ zbuf) {
  __shared__ alignas(16) unsigned short As[128 * LDSS];
  __shared__ alignas(16) unsigned short Bs[64 * LDSS];
  const int z  = blockIdx.z;
  const int b  = z / H_, h = z % H_;
  const int m0 = blockIdx.x << 7;
  const unsigned short* A  = alpha + (size_t)z * L_ * L_ + (size_t)m0 * L_;
  const unsigned short* Bp = VT + (size_t)z * DK_ * L_;
  auto epi = [=](int mt, int nt, const v8f& a) {
    const int lane = threadIdx.x & 31;
    const int row0 = m0 + mt + ((lane >> 4) << 3);
    const int col  = nt + (lane & 15);        // d in 0..63
#pragma unroll
    for (int v = 0; v < 8; ++v) {
      const int l = row0 + v;
      zbuf[((size_t)(b * L_ + l)) * F_ + h * DK_ + col] = f2bf(a[v]);
    }
  };
  gemm_tile(A, L_, Bp, L_, L_, As, Bs, epi);
}

// out = z @ WO^T + bO, f32 into d_out region 1.
__global__ void __launch_bounds__(256)
oproj_kernel(const unsigned short* __restrict__ Z, const unsigned short* __restrict__ W,
             const float* __restrict__ bias, float* __restrict__ out) {
  __shared__ alignas(16) unsigned short As[128 * LDSS];
  __shared__ alignas(16) unsigned short Bs[64 * LDSS];
  const int m0 = blockIdx.x << 7;
  const int n0 = blockIdx.y << 6;
  const unsigned short* A  = Z + (size_t)m0 * F_;
  const unsigned short* Bp = W + (size_t)n0 * F_;
  auto epi = [=](int mt, int nt, const v8f& a) {
    const int lane = threadIdx.x & 31;
    const int row0 = m0 + mt + ((lane >> 4) << 3);
    const int col  = n0 + nt + (lane & 15);
    const float bv = bias[col];
#pragma unroll
    for (int v = 0; v < 8; ++v)
      out[(size_t)(row0 + v) * F_ + col] = a[v] + bv;
  };
  gemm_tile(A, F_, Bp, F_, F_, As, Bs, epi);
}

// ------------------------------ launch -------------------------------------

extern "C" void kernel_launch(void* const* d_in, const int* in_sizes, int n_in,
                              void* d_out, int out_size, void* d_ws, size_t ws_size,
                              hipStream_t stream) {
  const float* qx  = (const float*)d_in[0];
  const float* kx  = (const float*)d_in[1];
  const float* vx  = (const float*)d_in[2];
  const float* pre = (const float*)d_in[3];
  const int*   msk = (const int*)d_in[4];
  const float* WQ  = (const float*)d_in[5];
  const float* bQ  = (const float*)d_in[6];
  const float* WK  = (const float*)d_in[7];
  const float* bK  = (const float*)d_in[8];
  const float* WV  = (const float*)d_in[9];
  const float* bV  = (const float*)d_in[10];
  const float* WO  = (const float*)d_in[11];
  const float* bO  = (const float*)d_in[12];
  (void)in_sizes; (void)n_in; (void)out_size; (void)ws_size;

  unsigned short* ws = (unsigned short*)d_ws;
  size_t o = 0;
  auto take = [&](size_t n) { unsigned short* p = ws + o; o += n; return p; };
  unsigned short* qxb = take((size_t)M_ * F_);
  unsigned short* kxb = take((size_t)M_ * F_);
  unsigned short* vxb = take((size_t)M_ * F_);
  unsigned short* wqb = take((size_t)F_ * F_);
  unsigned short* wkb = take((size_t)F_ * F_);
  unsigned short* wvb = take((size_t)F_ * F_);
  unsigned short* wob = take((size_t)F_ * F_);
  unsigned short* Qb  = take((size_t)M_ * F_);
  unsigned short* Kb  = take((size_t)M_ * F_);
  unsigned short* VTb = take((size_t)M_ * F_);
  unsigned short* alp = take((size_t)B_ * H_ * L_ * L_);
  unsigned short* zb  = take((size_t)M_ * F_);

  float* z_out = (float*)d_out;                       // (B, L, H*DK)
  float* s_out = z_out + (size_t)M_ * F_;             // (B, H, L, L)

  const int NXE = M_ * F_;
  const int NWE = F_ * F_;
  dim3 blk(256);

  cvt_kernel<<<(NXE + 255) / 256, blk, 0, stream>>>(qx, qxb, NXE);
  cvt_kernel<<<(NXE + 255) / 256, blk, 0, stream>>>(kx, kxb, NXE);
  cvt_kernel<<<(NXE + 255) / 256, blk, 0, stream>>>(vx, vxb, NXE);
  cvt_kernel<<<(NWE + 255) / 256, blk, 0, stream>>>(WQ, wqb, NWE);
  cvt_kernel<<<(NWE + 255) / 256, blk, 0, stream>>>(WK, wkb, NWE);
  cvt_kernel<<<(NWE + 255) / 256, blk, 0, stream>>>(WV, wvb, NWE);
  cvt_kernel<<<(NWE + 255) / 256, blk, 0, stream>>>(WO, wob, NWE);

  dim3 gproj(M_ / 128, F_ / 64);
  proj_kernel<<<gproj, blk, 0, stream>>>(qxb, wqb, bQ, Qb, 0);
  proj_kernel<<<gproj, blk, 0, stream>>>(kxb, wkb, bK, Kb, 0);
  proj_kernel<<<gproj, blk, 0, stream>>>(vxb, wvb, bV, VTb, 1);

  dim3 gqk(L_ / 128, L_ / 64, B_ * H_);
  qk_kernel<<<gqk, blk, 0, stream>>>(Qb, Kb, pre, msk, s_out);

  softmax_kernel<<<B_ * H_ * L_, blk, 0, stream>>>(s_out, alp);

  dim3 gav(L_ / 128, 1, B_ * H_);
  av_kernel<<<gav, blk, 0, stream>>>(alp, VTb, zb);

  dim3 gop(M_ / 128, F_ / 64);
  oproj_kernel<<<gop, blk, 0, stream>>>(zb, wob, bO, z_out);
}
